// EuclideanCodebook_84215718740327
// MI455X (gfx1250) — matrix operand order repaired and checked by reading, not compiled, not run on hardware
//
#include <hip/hip_runtime.h>
#include <hip/hip_bf16.h>

typedef __attribute__((ext_vector_type(16))) __bf16 v16bf;
typedef __attribute__((ext_vector_type(8)))  float  v8f;

#define N_ROWS   65536   // BATCH*SEQ
#define DIM      256
#define KCODES   1024
#define TILE_M   128
#define TILE_N   128
#define LDSS     36      // padded f32 row stride in LDS (kills 64-bank conflicts)

// Async global->LDS staging if the toolchain exposes the gfx1250 builtins.
#if defined(__has_builtin)
#if __has_builtin(__builtin_amdgcn_global_load_async_to_lds_b128) && \
    __has_builtin(__builtin_amdgcn_s_wait_asynccnt)
#define USE_ASYNC_LDS 1
#endif
#endif
#ifndef USE_ASYNC_LDS
#define USE_ASYNC_LDS 0
#endif

#if USE_ASYNC_LDS
// Parameter types per hipcc diagnostic: int4 (vector_size 16) pointers,
// param0 in the global/device address space, param1 in LDS.
typedef int v4i_vs __attribute__((vector_size(16)));
typedef __attribute__((address_space(1))) v4i_vs* async_gptr;
typedef __attribute__((address_space(3))) v4i_vs* async_lptr;
#endif

// ---------------------------------------------------------------------------
// Kernel 0: per-row squared norm (wave per row, 256 f32 per row)
// ---------------------------------------------------------------------------
__global__ __launch_bounds__(256) void vq_rowsq(const float* __restrict__ a,
                                                float* __restrict__ out, int rows)
{
    int gw   = (blockIdx.x * 256 + threadIdx.x) >> 5;   // global wave = row
    int lane = threadIdx.x & 31;
    if (gw >= rows) return;
    const float4* r = (const float4*)(a + (size_t)gw * DIM);
    float4 v0 = r[lane];
    float4 v1 = r[lane + 32];
    float s = v0.x*v0.x + v0.y*v0.y + v0.z*v0.z + v0.w*v0.w
            + v1.x*v1.x + v1.y*v1.y + v1.z*v1.z + v1.w*v1.w;
#pragma unroll
    for (int off = 16; off >= 1; off >>= 1)
        s += __shfl_xor(s, off, 32);
    if (lane == 0) out[gw] = s;
}

// ---------------------------------------------------------------------------
// Kernel 1: tiled bf16x3 WMMA GEMM -> dist_sq + per-64-col partial argmin
// grid = (KCODES/TILE_N, N_ROWS/TILE_M), block = 256 (8 wave32)
// ---------------------------------------------------------------------------
__global__ __launch_bounds__(256) void vq_gemm(const float* __restrict__ x,
                                               const float* __restrict__ embed,
                                               const float* __restrict__ xsq,
                                               const float* __restrict__ esq,
                                               float* __restrict__ dist_out,
                                               float* __restrict__ pdist,
                                               float* __restrict__ pidx)
{
    __shared__ float ldsX[TILE_M * LDSS];
    __shared__ float ldsE[TILE_N * LDSS];

    const int tid  = threadIdx.x;
    const int lane = tid & 31;
    const int wid  = tid >> 5;
    const int lrow = lane & 15;

    const int rowTile = blockIdx.y * TILE_M;     // over N rows
    const int colTile = blockIdx.x * TILE_N;     // over K codes

    const int rowHalfBase = (wid >> 1) * 32;     // 0,32,64,96
    const int colHalf     = wid & 1;
    const int colBase     = colHalf * 64;

    v8f acc[2][4];
#pragma unroll
    for (int i = 0; i < 2; i++)
#pragma unroll
        for (int j = 0; j < 4; j++)
            acc[i][j] = (v8f){0.f,0.f,0.f,0.f,0.f,0.f,0.f,0.f};

    for (int dc = 0; dc < DIM; dc += 32) {
        // ---- stage 128x32 f32 chunks of X and E into LDS -----------------
#if USE_ASYNC_LDS
        {
            // One async b128 instruction moves 32 lanes x 16B = 512B.
            // X chunk = 1024 float4s = 8 waves x 4 instr; same for E.
            int f0 = (wid * 4) * 32 + lane;
#pragma unroll
            for (int q = 0; q < 4; q++) {
                int fq  = f0 + q * 32;
                int row = fq >> 3;
                int v   = (fq & 7) * 4;
                __builtin_amdgcn_global_load_async_to_lds_b128(
                    (async_gptr)(x + (size_t)(rowTile + row) * DIM + dc + v),
                    (async_lptr)(&ldsX[row * LDSS + v]), 0, 0);
                __builtin_amdgcn_global_load_async_to_lds_b128(
                    (async_gptr)(embed + (size_t)(colTile + row) * DIM + dc + v),
                    (async_lptr)(&ldsE[row * LDSS + v]), 0, 0);
            }
            __builtin_amdgcn_s_wait_asynccnt(0);
        }
#else
#pragma unroll
        for (int p = 0; p < 4; p++) {
            int r = (tid >> 3) + 32 * p;
            int v = (tid & 7) * 4;
            *(float4*)&ldsX[r * LDSS + v] =
                *(const float4*)(x + (size_t)(rowTile + r) * DIM + dc + v);
            *(float4*)&ldsE[r * LDSS + v] =
                *(const float4*)(embed + (size_t)(colTile + r) * DIM + dc + v);
        }
        if (dc + 32 < DIM) {   // warm L2 for the next chunk (global_prefetch_b8)
            __builtin_prefetch(x + (size_t)(rowTile + (tid >> 3)) * DIM + dc + 32, 0, 3);
            __builtin_prefetch(embed + (size_t)(colTile + (tid >> 3)) * DIM + dc + 32, 0, 3);
        }
#endif
        __syncthreads();

        // ---- build B fragments (lanes 0-15: K 0-15; lanes 16-31: K 16-31)
        const int kB16 = (lane >> 4) * 16;
        v16bf bhi[4], blo[4];
#pragma unroll
        for (int j = 0; j < 4; j++) {
            float t[16];
            const float* src = &ldsE[(colBase + j * 16 + lrow) * LDSS + kB16];
            *(float4*)&t[0]  = *(const float4*)&src[0];
            *(float4*)&t[4]  = *(const float4*)&src[4];
            *(float4*)&t[8]  = *(const float4*)&src[8];
            *(float4*)&t[12] = *(const float4*)&src[12];
#pragma unroll
            for (int e = 0; e < 16; e++) {
                __bf16 h = (__bf16)t[e];
                bhi[j][e] = h;
                blo[j][e] = (__bf16)(t[e] - (float)h);
            }
        }

        // ---- A fragments (lanes 0-15: K 0-15; lanes 16-31: K 16-31) --
        const int kbA = (lane >> 4) * 16;
#pragma unroll
        for (int i = 0; i < 2; i++) {
            float t[16];
            const float* src = &ldsX[(rowHalfBase + i * 16 + lrow) * LDSS];
            *(float4*)&t[0]  = *(const float4*)&src[kbA + 0];
            *(float4*)&t[4]  = *(const float4*)&src[kbA + 4];
            *(float4*)&t[8]  = *(const float4*)&src[kbA + 8];
            *(float4*)&t[12] = *(const float4*)&src[kbA + 12];
            v16bf ahi, alo;
#pragma unroll
            for (int e = 0; e < 16; e++) {
                __bf16 h = (__bf16)t[e];
                ahi[e] = h;
                alo[e] = (__bf16)(t[e] - (float)h);
            }
            // Sweep j per split-term: consecutive WMMAs hit independent
            // accumulators, hiding the D->src hazard window without v_nops.
#pragma unroll
            for (int j = 0; j < 4; j++)
                acc[i][j] = __builtin_amdgcn_wmma_f32_16x16x32_bf16(
                    false, ahi, false, bhi[j], (short)0, acc[i][j], false, false);
#pragma unroll
            for (int j = 0; j < 4; j++)
                acc[i][j] = __builtin_amdgcn_wmma_f32_16x16x32_bf16(
                    false, ahi, false, blo[j], (short)0, acc[i][j], false, false);
#pragma unroll
            for (int j = 0; j < 4; j++)
                acc[i][j] = __builtin_amdgcn_wmma_f32_16x16x32_bf16(
                    false, alo, false, bhi[j], (short)0, acc[i][j], false, false);
        }
        __syncthreads();
    }

    // ---- epilogue: dist = |x|^2 - 2*dot + |e|^2, store + partial argmin --
    const int hi8 = (lane >> 4) * 8;   // C/D layout: lanes 16-31 hold M+8
    float xs[2][8];
#pragma unroll
    for (int i = 0; i < 2; i++)
#pragma unroll
        for (int r = 0; r < 8; r++)
            xs[i][r] = xsq[rowTile + rowHalfBase + i * 16 + hi8 + r];
    float es[4];
#pragma unroll
    for (int j = 0; j < 4; j++)
        es[j] = esq[colTile + colBase + j * 16 + lrow];

    float bestd[2][8];
    int   bestn[2][8];
#pragma unroll
    for (int i = 0; i < 2; i++)
#pragma unroll
        for (int r = 0; r < 8; r++) { bestd[i][r] = 3.4e38f; bestn[i][r] = 0x7fffffff; }

#pragma unroll
    for (int i = 0; i < 2; i++) {
#pragma unroll
        for (int j = 0; j < 4; j++) {
            int ng = colTile + colBase + j * 16 + lrow;
#pragma unroll
            for (int r = 0; r < 8; r++) {
                int mg = rowTile + rowHalfBase + i * 16 + hi8 + r;
                float dd = xs[i][r] - 2.0f * acc[i][j][r] + es[j];
                dist_out[(size_t)mg * KCODES + ng] = dd;
                if (dd < bestd[i][r]) { bestd[i][r] = dd; bestn[i][r] = ng; }
            }
        }
    }

    // 16-lane argmin reduction per row, tie-break on lower code index
#pragma unroll
    for (int i = 0; i < 2; i++) {
#pragma unroll
        for (int r = 0; r < 8; r++) {
            float d = bestd[i][r];
            int   n = bestn[i][r];
#pragma unroll
            for (int off = 8; off >= 1; off >>= 1) {
                float d2 = __shfl_xor(d, off, 16);
                int   n2 = __shfl_xor(n, off, 16);
                if (d2 < d || (d2 == d && n2 < n)) { d = d2; n = n2; }
            }
            if (lrow == 0) {
                int mg = rowTile + rowHalfBase + i * 16 + hi8 + r;
                int p  = blockIdx.x * 2 + colHalf;   // 16 partials per row
                pdist[(size_t)mg * 16 + p] = d;
                pidx [(size_t)mg * 16 + p] = (float)n;
            }
        }
    }
}

// ---------------------------------------------------------------------------
// Kernel 2: reduce 16 partials per row -> index, gather codebook row
// ---------------------------------------------------------------------------
__global__ __launch_bounds__(256) void vq_finalize(const float* __restrict__ embed,
                                                   const float* __restrict__ pdist,
                                                   const float* __restrict__ pidx,
                                                   float* __restrict__ quant,
                                                   float* __restrict__ ind_out,
                                                   float* __restrict__ nexp_out)
{
    int n    = (blockIdx.x * 256 + threadIdx.x) >> 5;   // wave per row
    int lane = threadIdx.x & 31;
    if (n >= N_ROWS) return;

    int p = lane & 15;   // both lane halves load identical data -> same result
    float d  = pdist[(size_t)n * 16 + p];
    float fi = pidx [(size_t)n * 16 + p];
#pragma unroll
    for (int off = 8; off >= 1; off >>= 1) {
        float d2 = __shfl_xor(d,  off, 16);
        float f2 = __shfl_xor(fi, off, 16);
        if (d2 < d || (d2 == d && f2 < fi)) { d = d2; fi = f2; }
    }
    int idx = (int)fi;
    if (lane == 0) ind_out[n] = fi;

    const float4* src = (const float4*)(embed + (size_t)idx * DIM);
    float4*       dst = (float4*)(quant + (size_t)n * DIM);
    dst[lane]      = src[lane];
    dst[lane + 32] = src[lane + 32];

    if (blockIdx.x == 0 && threadIdx.x == 0) nexp_out[0] = 0.0f;
}

// ---------------------------------------------------------------------------
extern "C" void kernel_launch(void* const* d_in, const int* in_sizes, int n_in,
                              void* d_out, int out_size, void* d_ws, size_t ws_size,
                              hipStream_t stream)
{
    const float* x     = (const float*)d_in[0];   // [8,8192,256] f32
    const float* embed = (const float*)d_in[1];   // [1024,256]  f32

    float* out   = (float*)d_out;
    float* quant = out;                               // 16,777,216
    float* ind   = out + (size_t)N_ROWS * DIM;        //     65,536
    float* nexp  = ind + N_ROWS;                      //          1
    float* dist  = nexp + 1;                          // 67,108,864

    float* xsq   = (float*)d_ws;                      // 65,536
    float* esq   = xsq + N_ROWS;                      //  1,024
    float* pd    = esq + KCODES;                      // 65,536*16
    float* pi    = pd + (size_t)N_ROWS * 16;          // 65,536*16

    vq_rowsq<<<N_ROWS / 8, 256, 0, stream>>>(x, xsq, N_ROWS);
    vq_rowsq<<<KCODES / 8, 256, 0, stream>>>(embed, esq, KCODES);
    vq_gemm<<<dim3(KCODES / TILE_N, N_ROWS / TILE_M), 256, 0, stream>>>(
        x, embed, xsq, esq, dist, pd, pi);
    vq_finalize<<<N_ROWS / 8, 256, 0, stream>>>(embed, pd, pi, quant, ind, nexp);
}
